// GNN_6047313952840
// MI455X (gfx1250) — compile-verified, matrix operand order checked by reading, not run on hardware
//
#include <hip/hip_runtime.h>
#include <hip/hip_bf16.h>
#include <math.h>

// Problem constants (match reference)
#define N_NODES 10000
#define N_EDGES 160000
#define F_DIM   512
#define N_CLS   16
#define M_PAD   10112        // 79 * 128 (rows padded for 128-row GEMM blocks)
#define LDSB_STRIDE 528      // padded LDS row stride (u16 elems) to reduce bank conflicts

typedef __attribute__((ext_vector_type(16))) __bf16 v16bf;
typedef __attribute__((ext_vector_type(8)))  float  v8f;

__device__ __forceinline__ unsigned short f2bf_rne(float f) {
  unsigned int u = __float_as_uint(f);
  unsigned int lsb = (u >> 16) & 1u;
  u += 0x7fffu + lsb;                 // round-to-nearest-even
  return (unsigned short)(u >> 16);
}

// ---------- conversion: f32 -> bf16 with zero padding past nValid ----------
__global__ __launch_bounds__(256)
void k_f32_to_bf16_pad(const float* __restrict__ src, unsigned short* __restrict__ dst,
                       long long nValid, long long nTotal) {
  long long i = (long long)blockIdx.x * blockDim.x + threadIdx.x;
  if (i < nTotal) dst[i] = (i < nValid) ? f2bf_rne(src[i]) : (unsigned short)0;
}

// ---------- degree / rsqrt ----------
__global__ __launch_bounds__(256)
void k_deg_init(float* __restrict__ deg, int n) {
  int i = blockIdx.x * blockDim.x + threadIdx.x;
  if (i < n) deg[i] = 1.0f;                 // self-loop contribution
}
__global__ __launch_bounds__(256)
void k_deg_count(const long long* __restrict__ ei, float* __restrict__ deg, int E) {
  int e = blockIdx.x * blockDim.x + threadIdx.x;
  if (e < E) atomicAdd(&deg[(int)ei[(size_t)E + e]], 1.0f);   // dst row of edge_index
}
__global__ __launch_bounds__(256)
void k_rsqrt_inplace(float* __restrict__ d, int n) {
  int i = blockIdx.x * blockDim.x + threadIdx.x;
  if (i < n) d[i] = rsqrtf(d[i]);
}

// ---------- bf16 WMMA GEMM: C[M_PAD,512] = A[M_PAD,512] * B[512,512] ----------
// Block: 256 threads = 8 waves, computes a 128(M) x 32(N) strip.
// B panel (512 x 32) staged transposed in LDS; each wave owns TWO 16x16 C tiles
// (same M rows, adjacent N tiles) so every A fragment feeds two WMMAs.
// K is a compile-time constant so the ping-pong double buffer fully unrolls
// and lives entirely in VGPRs (no dynamic indexing, no scratch spills).
__global__ __launch_bounds__(256)
void k_gemm_bf16(const unsigned short* __restrict__ A,
                 const unsigned short* __restrict__ B,
                 float* __restrict__ C) {
  constexpr int K  = F_DIM;   // 512
  constexpr int Nb = F_DIM;   // 512
  constexpr int STEPS = K / 32;   // 16

  __shared__ unsigned short ldsB[32 * LDSB_STRIDE];

  const int n0 = blockIdx.y * 32;
  const int m0 = blockIdx.x * 128;
  const int tid = threadIdx.x;

  // Cooperative transposed staging: ldsB[n][k] = B[k][n0+n], two columns per u32 load.
  for (int idx = tid; idx < K * 16; idx += 256) {
    int k  = idx >> 4;
    int np = (idx & 15) << 1;
    unsigned int v = *(const unsigned int*)(B + (size_t)k * Nb + n0 + np);
    ldsB[np * LDSB_STRIDE + k]       = (unsigned short)(v & 0xffffu);
    ldsB[(np + 1) * LDSB_STRIDE + k] = (unsigned short)(v >> 16);
  }
  __syncthreads();

  const int wave = tid >> 5;
  const int lane = tid & 31;
  const int hi   = lane >> 4;          // K-half selector per ISA A/B fragment layout
  const int mrow = m0 + wave * 16 + (lane & 15);

  const unsigned int* Arow  = (const unsigned int*)(A + (size_t)mrow * K);
  const unsigned int* Bcol0 = (const unsigned int*)(ldsB + (lane & 15) * LDSB_STRIDE);
  const unsigned int* Bcol1 = (const unsigned int*)(ldsB + (16 + (lane & 15)) * LDSB_STRIDE);

  union Frag { v16bf v; unsigned int d[8]; };

  auto ldFrag = [](Frag& f, const unsigned int* p, int base) {
#pragma unroll
    for (int q = 0; q < 4; ++q) {
      f.d[q]     = p[base + q];          // K = base*2 + 2q .. +1
      f.d[q + 4] = p[base + 8 + q];      // K = base*2 + 16 + 2q .. +1
    }
  };

  const int kbase = hi * 4;              // dword base for this lane's K-half

  // Ping-pong register sets (no arrays -> guaranteed SSA promotion).
  Frag a0, p0, q0, a1, p1, q1;
  ldFrag(a0, Arow,  kbase);
  ldFrag(p0, Bcol0, kbase);
  ldFrag(q0, Bcol1, kbase);

  v8f acc0 = {}, acc1 = {};
#pragma unroll
  for (int s = 0; s < STEPS; ++s) {
    if (s + 1 < STEPS) {                 // compile-time under full unroll
      const int nb = kbase + (s + 1) * 16;
      __builtin_prefetch(Arow + nb + 32, 0, 3);   // A stream, ~2 K-steps ahead
      ldFrag(a1, Arow,  nb);
      ldFrag(p1, Bcol0, nb);
      ldFrag(q1, Bcol1, nb);
    }
    acc0 = __builtin_amdgcn_wmma_f32_16x16x32_bf16(
        false, a0.v, false, p0.v, (short)0, acc0, false, false);
    acc1 = __builtin_amdgcn_wmma_f32_16x16x32_bf16(
        false, a0.v, false, q0.v, (short)0, acc1, false, false);
    a0 = a1; p0 = p1; q0 = q1;           // register rotation, folded by unroller
  }

  // C/D layout: VGPR r -> row hi*8 + r, col = lane&15
  float* Cp0 = C + (size_t)(m0 + wave * 16 + hi * 8) * Nb + n0 + (lane & 15);
  float* Cp1 = Cp0 + 16;
#pragma unroll
  for (int r = 0; r < 8; ++r) {
    Cp0[(size_t)r * Nb] = acc0[r];
    Cp1[(size_t)r * Nb] = acc1[r];
  }
}

// ---------- aggregation: agg = D^-1/2 (A+I) D^-1/2 h ----------
__global__ __launch_bounds__(256)
void k_self_loop_init(const float* __restrict__ h, const float* __restrict__ dinv,
                      float* __restrict__ agg, int nTot) {
  int i = blockIdx.x * blockDim.x + threadIdx.x;
  if (i < nTot) {
    int node = i >> 9;                    // F_DIM == 512
    float di = dinv[node];
    agg[i] = di * di * h[i];
  }
}
__global__ __launch_bounds__(256)
void k_agg_edges(const long long* __restrict__ ei, const float* __restrict__ dinv,
                 const float* __restrict__ h, float* __restrict__ agg, int E) {
  long long idx = (long long)blockIdx.x * blockDim.x + threadIdx.x;
  if (idx < (long long)E * (F_DIM / 4)) {
    int e  = (int)(idx >> 7);             // F_DIM/4 == 128
    int fo = ((int)idx & 127) << 2;
    int s = (int)ei[e];
    int d = (int)ei[(size_t)E + e];
    float nrm = dinv[s] * dinv[d];
    float4 v = *(const float4*)(h + (size_t)s * F_DIM + fo);
    float* ap = agg + (size_t)d * F_DIM + fo;
    atomicAdd(ap + 0, nrm * v.x);
    atomicAdd(ap + 1, nrm * v.y);
    atomicAdd(ap + 2, nrm * v.z);
    atomicAdd(ap + 3, nrm * v.w);
  }
}

// ---------- bias + ReLU ----------
__global__ __launch_bounds__(256)
void k_bias_relu_bf16(const float* __restrict__ agg, const float* __restrict__ bias,
                      unsigned short* __restrict__ out, int nTot) {
  int i = blockIdx.x * blockDim.x + threadIdx.x;
  if (i < nTot) {
    float v = agg[i] + bias[i & (F_DIM - 1)];
    out[i] = f2bf_rne(v > 0.0f ? v : 0.0f);
  }
}
__global__ __launch_bounds__(256)
void k_bias_relu_f32(const float* __restrict__ agg, const float* __restrict__ bias,
                     float* __restrict__ out, int nTot) {
  int i = blockIdx.x * blockDim.x + threadIdx.x;
  if (i < nTot) {
    float v = agg[i] + bias[i & (F_DIM - 1)];
    out[i] = v > 0.0f ? v : 0.0f;
  }
}

// ---------- fused output layer + softmax (one thread per node) ----------
__global__ __launch_bounds__(256)
void k_out_softmax(const float* __restrict__ h, const float* __restrict__ Wout,
                   const float* __restrict__ bout, float* __restrict__ out, int n) {
  int t = blockIdx.x * blockDim.x + threadIdx.x;
  if (t >= n) return;
  float acc[N_CLS];
#pragma unroll
  for (int c = 0; c < N_CLS; ++c) acc[c] = bout[c];
  const float* hr = h + (size_t)t * F_DIM;
  for (int k = 0; k < F_DIM; ++k) {
    float hv = hr[k];
    const float4* wr = (const float4*)(Wout + (size_t)k * N_CLS);
#pragma unroll
    for (int q = 0; q < 4; ++q) {
      float4 w = wr[q];
      acc[q * 4 + 0] += hv * w.x;
      acc[q * 4 + 1] += hv * w.y;
      acc[q * 4 + 2] += hv * w.z;
      acc[q * 4 + 3] += hv * w.w;
    }
  }
  float mx = acc[0];
#pragma unroll
  for (int c = 1; c < N_CLS; ++c) mx = fmaxf(mx, acc[c]);
  float s = 0.0f;
#pragma unroll
  for (int c = 0; c < N_CLS; ++c) { acc[c] = __expf(acc[c] - mx); s += acc[c]; }
  float inv = 1.0f / s;
#pragma unroll
  for (int c = 0; c < N_CLS; ++c) out[(size_t)t * N_CLS + c] = acc[c] * inv;
}

extern "C" void kernel_launch(void* const* d_in, const int* in_sizes, int n_in,
                              void* d_out, int out_size, void* d_ws, size_t ws_size,
                              hipStream_t stream) {
  (void)in_sizes; (void)n_in; (void)out_size; (void)ws_size;

  const float*     x    = (const float*)d_in[0];
  const long long* ei   = (const long long*)d_in[1];   // int64 [2, E]: row0 = src, row1 = dst
  const float*     W1   = (const float*)d_in[2];
  const float*     b1   = (const float*)d_in[3];
  const float*     W2   = (const float*)d_in[4];
  const float*     b2   = (const float*)d_in[5];
  const float*     Wout = (const float*)d_in[6];
  const float*     bout = (const float*)d_in[7];
  float*           out  = (float*)d_out;

  // Workspace layout (256B-aligned slices)
  char* ws = (char*)d_ws;
  size_t off = 0;
  auto take = [&](size_t bytes) -> char* {
    char* p = ws + off;
    off += (bytes + 255) & ~(size_t)255;
    return p;
  };
  unsigned short* xb   = (unsigned short*)take((size_t)M_PAD * F_DIM * 2); // bf16 activations
  unsigned short* w1b  = (unsigned short*)take((size_t)F_DIM * F_DIM * 2);
  unsigned short* w2b  = (unsigned short*)take((size_t)F_DIM * F_DIM * 2);
  float*          h    = (float*)take((size_t)M_PAD * F_DIM * 4);          // GEMM output
  float*          agg  = (float*)take((size_t)M_PAD * F_DIM * 4);          // aggregation
  float*          dinv = (float*)take((size_t)N_NODES * 4);                // deg -> rsqrt in place

  const long long nXTot = (long long)M_PAD * F_DIM;
  const long long nXVal = (long long)N_NODES * F_DIM;
  const long long nW    = (long long)F_DIM * F_DIM;
  const int nNF = N_NODES * F_DIM;

  dim3 blk(256);
  dim3 gConvX((unsigned)((nXTot + 255) / 256));
  dim3 gConvW((unsigned)((nW + 255) / 256));
  dim3 gNodes((N_NODES + 255) / 256);
  dim3 gEdges((N_EDGES + 255) / 256);
  dim3 gNF((nNF + 255) / 256);
  dim3 gEdgeF((unsigned)(((long long)N_EDGES * (F_DIM / 4) + 255) / 256));
  dim3 gGemm(M_PAD / 128, F_DIM / 32);   // 128-row x 32-col strips

  // Prep: conversions + normalization coefficients
  k_f32_to_bf16_pad<<<gConvX, blk, 0, stream>>>(x, xb, nXVal, nXTot);
  k_f32_to_bf16_pad<<<gConvW, blk, 0, stream>>>(W1, w1b, nW, nW);
  k_f32_to_bf16_pad<<<gConvW, blk, 0, stream>>>(W2, w2b, nW, nW);
  k_deg_init<<<gNodes, blk, 0, stream>>>(dinv, N_NODES);
  k_deg_count<<<gEdges, blk, 0, stream>>>(ei, dinv, N_EDGES);
  k_rsqrt_inplace<<<gNodes, blk, 0, stream>>>(dinv, N_NODES);

  // Layer 1: h = xb @ W1 ; agg = norm-aggregate(h) ; xb = bf16(relu(agg + b1))
  k_gemm_bf16<<<gGemm, blk, 0, stream>>>(xb, w1b, h);
  k_self_loop_init<<<gNF, blk, 0, stream>>>(h, dinv, agg, nNF);
  k_agg_edges<<<gEdgeF, blk, 0, stream>>>(ei, dinv, h, agg, N_EDGES);
  k_bias_relu_bf16<<<gNF, blk, 0, stream>>>(agg, b1, xb, nNF);

  // Layer 2: h = xb @ W2 ; agg = norm-aggregate(h) ; h = relu(agg + b2)
  k_gemm_bf16<<<gGemm, blk, 0, stream>>>(xb, w2b, h);
  k_self_loop_init<<<gNF, blk, 0, stream>>>(h, dinv, agg, nNF);
  k_agg_edges<<<gEdgeF, blk, 0, stream>>>(ei, dinv, h, agg, N_EDGES);
  k_bias_relu_f32<<<gNF, blk, 0, stream>>>(agg, b2, h, nNF);

  // Output layer + softmax
  k_out_softmax<<<gNodes, blk, 0, stream>>>(h, Wout, bout, out, N_NODES);
}
